// QLoRALinear_31705448579867
// MI455X (gfx1250) — matrix operand order repaired and checked by reading, not compiled, not run on hardware
//
#include <hip/hip_runtime.h>

typedef __attribute__((ext_vector_type(16))) _Float16 v16h;
typedef __attribute__((ext_vector_type(8)))  _Float16 v8h;
typedef __attribute__((ext_vector_type(8)))  float    v8f;

#define BM 128
#define BN 128
#define BK 64          // == GROUP_SIZE
#define N_TOTAL 4096
#define K_TOTAL 4096
#define NGROUPS (K_TOTAL / BK)
#define RANK 16

// ---------------------------------------------------------------------------
// async copy helper: 64 contiguous bytes global -> LDS per lane, 4x b128.
// INST_OFFSET is added to BOTH the LDS and global addresses (ISA 08 §4.4).
// ---------------------------------------------------------------------------
__device__ __forceinline__ void async_copy64(unsigned lds_off, const void* gaddr) {
  asm volatile(
      "global_load_async_to_lds_b128 %0, %1, off\n\t"
      "global_load_async_to_lds_b128 %0, %1, off offset:16\n\t"
      "global_load_async_to_lds_b128 %0, %1, off offset:32\n\t"
      "global_load_async_to_lds_b128 %0, %1, off offset:48"
      :: "v"(lds_off), "v"(gaddr) : "memory");
}

__device__ __forceinline__ void wait_async0() {
  asm volatile("s_wait_asynccnt 0x0" ::: "memory");
}

// ---------------------------------------------------------------------------
// Prepass 1: x (fp32) -> x16 (f16), 8 elements / thread
// ---------------------------------------------------------------------------
__global__ __launch_bounds__(256)
void qlora_cvt_x(const float* __restrict__ x, _Float16* __restrict__ x16) {
  const size_t i = ((size_t)blockIdx.x * 256 + threadIdx.x) * 8;
  const float4* s = (const float4*)(x + i);
  float4 f0 = s[0], f1 = s[1];
  v8h h = {(_Float16)f0.x, (_Float16)f0.y, (_Float16)f0.z, (_Float16)f0.w,
           (_Float16)f1.x, (_Float16)f1.y, (_Float16)f1.z, (_Float16)f1.w};
  *(v8h*)(x16 + i) = h;
}

// ---------------------------------------------------------------------------
// Prepass 2: dequant 4-bit codes -> w16 (f16), 8 elements / thread
// ---------------------------------------------------------------------------
__global__ __launch_bounds__(256)
void qlora_dequant_w(const int* __restrict__ qw, const float* __restrict__ scales,
                     const float* __restrict__ zeros, _Float16* __restrict__ w16) {
  const size_t i = ((size_t)blockIdx.x * 256 + threadIdx.x) * 8;
  const int n = (int)(i >> 12);            // / K_TOTAL
  const int k = (int)(i & (K_TOTAL - 1));
  const int g = k >> 6;                    // / GROUP_SIZE
  const float sc  = scales[(size_t)n * NGROUPS + g];
  const float nzs = -zeros[(size_t)n * NGROUPS + g] * sc;
  const int4* s = (const int4*)(qw + i);
  int4 q0 = s[0], q1 = s[1];
  v8h h = {(_Float16)fmaf((float)q0.x, sc, nzs), (_Float16)fmaf((float)q0.y, sc, nzs),
           (_Float16)fmaf((float)q0.z, sc, nzs), (_Float16)fmaf((float)q0.w, sc, nzs),
           (_Float16)fmaf((float)q1.x, sc, nzs), (_Float16)fmaf((float)q1.y, sc, nzs),
           (_Float16)fmaf((float)q1.z, sc, nzs), (_Float16)fmaf((float)q1.w, sc, nzs)};
  *(v8h*)(w16 + i) = h;
}

// ---------------------------------------------------------------------------
// Prepass 3: u[m][r] = (alpha/rank) * dot(x[m,:], lora_A[r,:])
// ---------------------------------------------------------------------------
__global__ __launch_bounds__(256)
void qlora_lora_u(const float* __restrict__ x, const float* __restrict__ loraA,
                  float* __restrict__ u, int M) {
  const int idx = blockIdx.x * blockDim.x + threadIdx.x;   // m*16 + r
  const int m = idx >> 4;
  const int r = idx & 15;
  if (m >= M) return;
  const float4* xr = (const float4*)(x + (size_t)m * K_TOTAL);
  const float4* ar = (const float4*)(loraA + (size_t)r * K_TOTAL);
  float s = 0.0f;
  #pragma unroll 4
  for (int i = 0; i < K_TOTAL / 4; ++i) {
    float4 a = xr[i];
    float4 b = ar[i];
    s = fmaf(a.x, b.x, s); s = fmaf(a.y, b.y, s);
    s = fmaf(a.z, b.z, s); s = fmaf(a.w, b.w, s);
  }
  u[idx] = s * (1.0f / (float)RANK);
}

// ---------------------------------------------------------------------------
// Main GEMM (async path): y = x16 @ w16^T + u @ loraB^T + bias
// 128x128 tile / 256-thread block; 8 waves (4x2), 32x64 per wave.
// Double-buffered LDS (4 statically-addressed arrays), staged purely with
// global_load_async_to_lds_b128; pipeline loop unrolled by 2 so each half
// uses compile-time LDS bases (keeps accumulators pinned, no v_mov shuffles).
// ---------------------------------------------------------------------------
__global__ __launch_bounds__(256, 1)
void qlora_gemm_async(const _Float16* __restrict__ x16,
                      const _Float16* __restrict__ w16,
                      const float* __restrict__ loraB,
                      const float* __restrict__ bias,
                      const float* __restrict__ u,
                      float* __restrict__ out) {
  __shared__ __align__(16) _Float16 ldsX0[BM * BK];
  __shared__ __align__(16) _Float16 ldsX1[BM * BK];
  __shared__ __align__(16) _Float16 ldsW0[BN * BK];
  __shared__ __align__(16) _Float16 ldsW1[BN * BK];

  const int tid    = threadIdx.x;
  const int lane   = tid & 31;
  const int wave   = tid >> 5;
  const int lane16 = lane & 15;
  const int laneHi = lane >> 4;
  const int waveM  = wave >> 1;          // 0..3
  const int waveN  = wave & 1;           // 0..1

  const int tileN = blockIdx.x * BN;
  const int tileM = blockIdx.y * BM;

  v8f acc[2][4];
  const v8f vzero = {0.0f, 0.0f, 0.0f, 0.0f, 0.0f, 0.0f, 0.0f, 0.0f};
  #pragma unroll
  for (int i = 0; i < 2; ++i)
    #pragma unroll
    for (int j = 0; j < 4; ++j) acc[i][j] = vzero;

  // Staging map: thread t owns 64 bytes = 32 f16: row r = t>>1, cols h*32..+32
  const int sRow = tid >> 1;
  const int sCol = (tid & 1) * 32;
  const unsigned ldsOffX0 = (unsigned)(size_t)(&ldsX0[sRow * BK + sCol]);
  const unsigned ldsOffX1 = (unsigned)(size_t)(&ldsX1[sRow * BK + sCol]);
  const unsigned ldsOffW0 = (unsigned)(size_t)(&ldsW0[sRow * BK + sCol]);
  const unsigned ldsOffW1 = (unsigned)(size_t)(&ldsW1[sRow * BK + sCol]);
  const _Float16* gx = x16 + (size_t)(tileM + sRow) * K_TOTAL + sCol;
  const _Float16* gw = w16 + (size_t)(tileN + sRow) * K_TOTAL + sCol;

  auto wmma_step = [&](const _Float16* bx, const _Float16* bw, int kk) {
    v16h a[2], b[4];
    #pragma unroll
    for (int fm = 0; fm < 2; ++fm) {
      // A (16-bit 16x32): m = lane%16; K chunks [laneHi*8..+8), [16+laneHi*8..+8)
      const _Float16* base = bx + (size_t)(waveM * 32 + fm * 16 + lane16) * BK + kk;
      v8h lo = *(const v8h*)(base + laneHi * 8);
      v8h hi = *(const v8h*)(base + 16 + laneHi * 8);
      a[fm] = __builtin_shufflevector(lo, hi, 0, 1, 2, 3, 4, 5, 6, 7,
                                      8, 9, 10, 11, 12, 13, 14, 15);
    }
    #pragma unroll
    for (int fn = 0; fn < 4; ++fn) {
      // B (16-bit 32x16): n = lane%16; K chunks [laneHi*8..+8), [16+laneHi*8..+8)
      const _Float16* base =
          bw + (size_t)(waveN * 64 + fn * 16 + lane16) * BK + kk + laneHi * 8;
      v8h lo = *(const v8h*)(base);
      v8h hi = *(const v8h*)(base + 16);
      b[fn] = __builtin_shufflevector(lo, hi, 0, 1, 2, 3, 4, 5, 6, 7,
                                      8, 9, 10, 11, 12, 13, 14, 15);
    }
    #pragma unroll
    for (int fm = 0; fm < 2; ++fm)
      #pragma unroll
      for (int fn = 0; fn < 4; ++fn)
        acc[fm][fn] = __builtin_amdgcn_wmma_f32_16x16x32_f16(
            false, a[fm], false, b[fn], (short)0, acc[fm][fn], false, false);
  };

  // Pipeline unrolled x2: even groups live in buffer 0, odd groups in buffer 1.
  async_copy64(ldsOffX0, gx);
  async_copy64(ldsOffW0, gw);
  for (int g = 0; g < NGROUPS; g += 2) {
    // ---- even group g: compute from buffer 0, prefetch g+1 into buffer 1 ----
    wait_async0();
    __syncthreads();
    if (g + 1 < NGROUPS) {
      async_copy64(ldsOffX1, gx + (size_t)(g + 1) * BK);
      async_copy64(ldsOffW1, gw + (size_t)(g + 1) * BK);
    }
    wmma_step(ldsX0, ldsW0, 0);
    wmma_step(ldsX0, ldsW0, 32);

    // ---- odd group g+1: compute from buffer 1, prefetch g+2 into buffer 0 ----
    wait_async0();
    __syncthreads();
    if (g + 2 < NGROUPS) {
      async_copy64(ldsOffX0, gx + (size_t)(g + 2) * BK);
      async_copy64(ldsOffW0, gw + (size_t)(g + 2) * BK);
    }
    wmma_step(ldsX1, ldsW1, 0);
    wmma_step(ldsX1, ldsW1, 32);
  }

  // ---- LoRA tail: extra K=32 step, A <- u (pad 16->32), B <- loraB ----
  // Last compute read buffer 1; buffer 0 has no readers after the final
  // barrier, so it is safe to restage it here.
  {
    if ((tid & 1) == 0) {
      const float4* us = (const float4*)(u + (size_t)(tileM + sRow) * RANK);
      const float4* bs = (const float4*)(loraB + (size_t)(tileN + sRow) * RANK);
      _Float16* dx = &ldsX0[sRow * BK];
      _Float16* dw = &ldsW0[sRow * BK];
      #pragma unroll
      for (int v = 0; v < 2; ++v) {
        float4 f0 = us[2 * v + 0];
        float4 f1 = us[2 * v + 1];
        v8h hx = {(_Float16)f0.x, (_Float16)f0.y, (_Float16)f0.z, (_Float16)f0.w,
                  (_Float16)f1.x, (_Float16)f1.y, (_Float16)f1.z, (_Float16)f1.w};
        *(v8h*)(dx + 8 * v) = hx;
        float4 g0 = bs[2 * v + 0];
        float4 g1 = bs[2 * v + 1];
        v8h hw = {(_Float16)g0.x, (_Float16)g0.y, (_Float16)g0.z, (_Float16)g0.w,
                  (_Float16)g1.x, (_Float16)g1.y, (_Float16)g1.z, (_Float16)g1.w};
        *(v8h*)(dw + 8 * v) = hw;
      }
    } else {
      const _Float16 z0 = (_Float16)0.0f;
      v8h z = {z0, z0, z0, z0, z0, z0, z0, z0};
      *(v8h*)(&ldsX0[sRow * BK + 16]) = z;
      *(v8h*)(&ldsX0[sRow * BK + 24]) = z;
      *(v8h*)(&ldsW0[sRow * BK + 16]) = z;
      *(v8h*)(&ldsW0[sRow * BK + 24]) = z;
    }
  }
  __syncthreads();
  wmma_step(ldsX0, ldsW0, 0);

  // ---- epilogue: add bias, store ----
  #pragma unroll
  for (int fn = 0; fn < 4; ++fn) {
    const int n = tileN + waveN * 64 + fn * 16 + lane16;
    const float bv = bias[n];
    #pragma unroll
    for (int fm = 0; fm < 2; ++fm) {
      const int mBase = tileM + waveM * 32 + fm * 16 + laneHi * 8;
      #pragma unroll
      for (int r = 0; r < 8; ++r) {
        out[(size_t)(mBase + r) * N_TOTAL + n] = acc[fm][fn][r] + bv;
      }
    }
  }
}

// ---------------------------------------------------------------------------
// Fallback fused GEMM (round-1 style): only needs 512 KB workspace.
// ---------------------------------------------------------------------------
__global__ __launch_bounds__(256, 1)
void qlora_gemm_fused(const float* __restrict__ x,
                      const int*   __restrict__ qw,
                      const float* __restrict__ scales,
                      const float* __restrict__ zeros,
                      const float* __restrict__ loraB,
                      const float* __restrict__ bias,
                      const float* __restrict__ u,
                      float* __restrict__ out) {
  __shared__ __align__(16) _Float16 ldsX[BM * BK];
  __shared__ __align__(16) _Float16 ldsW[BN * BK];

  const int tid    = threadIdx.x;
  const int lane   = tid & 31;
  const int wave   = tid >> 5;
  const int lane16 = lane & 15;
  const int laneHi = lane >> 4;
  const int waveM  = wave >> 1;
  const int waveN  = wave & 1;
  const int tileN = blockIdx.x * BN;
  const int tileM = blockIdx.y * BM;

  v8f acc[2][4];
  const v8f vzero = {0.0f, 0.0f, 0.0f, 0.0f, 0.0f, 0.0f, 0.0f, 0.0f};
  #pragma unroll
  for (int i = 0; i < 2; ++i)
    #pragma unroll
    for (int j = 0; j < 4; ++j) acc[i][j] = vzero;

  const int sRow = tid >> 1;
  const int sCol = (tid & 1) * 32;

  auto wmma_step = [&](int kk) {
    v16h a[2], b[4];
    #pragma unroll
    for (int fm = 0; fm < 2; ++fm) {
      const _Float16* base = ldsX + (size_t)(waveM * 32 + fm * 16 + lane16) * BK + kk;
      v8h lo = *(const v8h*)(base + laneHi * 8);
      v8h hi = *(const v8h*)(base + 16 + laneHi * 8);
      a[fm] = __builtin_shufflevector(lo, hi, 0, 1, 2, 3, 4, 5, 6, 7,
                                      8, 9, 10, 11, 12, 13, 14, 15);
    }
    #pragma unroll
    for (int fn = 0; fn < 4; ++fn) {
      const _Float16* base =
          ldsW + (size_t)(waveN * 64 + fn * 16 + lane16) * BK + kk + laneHi * 8;
      v8h lo = *(const v8h*)(base);
      v8h hi = *(const v8h*)(base + 16);
      b[fn] = __builtin_shufflevector(lo, hi, 0, 1, 2, 3, 4, 5, 6, 7,
                                      8, 9, 10, 11, 12, 13, 14, 15);
    }
    #pragma unroll
    for (int fm = 0; fm < 2; ++fm)
      #pragma unroll
      for (int fn = 0; fn < 4; ++fn)
        acc[fm][fn] = __builtin_amdgcn_wmma_f32_16x16x32_f16(
            false, a[fm], false, b[fn], (short)0, acc[fm][fn], false, false);
  };

  for (int g = 0; g < NGROUPS; ++g) {
    const int kk0 = g * BK;
    __syncthreads();
    {
      const float4* src =
          (const float4*)(x + (size_t)(tileM + sRow) * K_TOTAL + kk0 + sCol);
      _Float16* dst = ldsX + (size_t)sRow * BK + sCol;
      #pragma unroll
      for (int v = 0; v < 4; ++v) {
        float4 f0 = src[2 * v + 0];
        float4 f1 = src[2 * v + 1];
        v8h h = {(_Float16)f0.x, (_Float16)f0.y, (_Float16)f0.z, (_Float16)f0.w,
                 (_Float16)f1.x, (_Float16)f1.y, (_Float16)f1.z, (_Float16)f1.w};
        *(v8h*)(dst + 8 * v) = h;
      }
    }
    {
      const int n = tileN + sRow;
      const float sc  = scales[(size_t)n * NGROUPS + g];
      const float nzs = -zeros[(size_t)n * NGROUPS + g] * sc;
      const int4* src = (const int4*)(qw + (size_t)n * K_TOTAL + kk0 + sCol);
      _Float16* dst = ldsW + (size_t)sRow * BK + sCol;
      #pragma unroll
      for (int v = 0; v < 4; ++v) {
        int4 q0 = src[2 * v + 0];
        int4 q1 = src[2 * v + 1];
        v8h h = {(_Float16)fmaf((float)q0.x, sc, nzs),
                 (_Float16)fmaf((float)q0.y, sc, nzs),
                 (_Float16)fmaf((float)q0.z, sc, nzs),
                 (_Float16)fmaf((float)q0.w, sc, nzs),
                 (_Float16)fmaf((float)q1.x, sc, nzs),
                 (_Float16)fmaf((float)q1.y, sc, nzs),
                 (_Float16)fmaf((float)q1.z, sc, nzs),
                 (_Float16)fmaf((float)q1.w, sc, nzs)};
        *(v8h*)(dst + 8 * v) = h;
      }
    }
    __syncthreads();
    wmma_step(0);
    wmma_step(32);
  }

  __syncthreads();
  {
    if ((tid & 1) == 0) {
      const float4* us = (const float4*)(u + (size_t)(tileM + sRow) * RANK);
      const float4* bs = (const float4*)(loraB + (size_t)(tileN + sRow) * RANK);
      _Float16* dx = ldsX + (size_t)sRow * BK;
      _Float16* dw = ldsW + (size_t)sRow * BK;
      #pragma unroll
      for (int v = 0; v < 2; ++v) {
        float4 f0 = us[2 * v + 0];
        float4 f1 = us[2 * v + 1];
        v8h hx = {(_Float16)f0.x, (_Float16)f0.y, (_Float16)f0.z, (_Float16)f0.w,
                  (_Float16)f1.x, (_Float16)f1.y, (_Float16)f1.z, (_Float16)f1.w};
        *(v8h*)(dx + 8 * v) = hx;
        float4 g0 = bs[2 * v + 0];
        float4 g1 = bs[2 * v + 1];
        v8h hw = {(_Float16)g0.x, (_Float16)g0.y, (_Float16)g0.z, (_Float16)g0.w,
                  (_Float16)g1.x, (_Float16)g1.y, (_Float16)g1.z, (_Float16)g1.w};
        *(v8h*)(dw + 8 * v) = hw;
      }
    } else {
      const _Float16 z0 = (_Float16)0.0f;
      v8h z = {z0, z0, z0, z0, z0, z0, z0, z0};
      *(v8h*)(ldsX + (size_t)sRow * BK + 16) = z;
      *(v8h*)(ldsX + (size_t)sRow * BK + 24) = z;
      *(v8h*)(ldsW + (size_t)sRow * BK + 16) = z;
      *(v8h*)(ldsW + (size_t)sRow * BK + 24) = z;
    }
  }
  __syncthreads();
  wmma_step(0);

  #pragma unroll
  for (int fn = 0; fn < 4; ++fn) {
    const int n = tileN + waveN * 64 + fn * 16 + lane16;
    const float bv = bias[n];
    #pragma unroll
    for (int fm = 0; fm < 2; ++fm) {
      const int mBase = tileM + waveM * 32 + fm * 16 + laneHi * 8;
      #pragma unroll
      for (int r = 0; r < 8; ++r) {
        out[(size_t)(mBase + r) * N_TOTAL + n] = acc[fm][fn][r] + bv;
      }
    }
  }
}

// ---------------------------------------------------------------------------
extern "C" void kernel_launch(void* const* d_in, const int* in_sizes, int n_in,
                              void* d_out, int out_size, void* d_ws, size_t ws_size,
                              hipStream_t stream) {
  const float* x      = (const float*)d_in[0];
  const int*   qw     = (const int*)d_in[1];
  const float* scales = (const float*)d_in[2];
  const float* zeros  = (const float*)d_in[3];
  const float* loraA  = (const float*)d_in[4];
  const float* loraB  = (const float*)d_in[5];
  const float* bias   = (const float*)d_in[6];
  float* out = (float*)d_out;

  const int M = in_sizes[0] / K_TOTAL;        // B*S = 8192
  const size_t xBytes = (size_t)M * K_TOTAL * sizeof(_Float16);
  const size_t wBytes = (size_t)N_TOTAL * K_TOTAL * sizeof(_Float16);
  const size_t uBytes = (size_t)M * RANK * sizeof(float);

  dim3 blk(256);
  dim3 gGemm(N_TOTAL / BN, (unsigned)(M / BM));

  if (ws_size >= xBytes + wBytes + uBytes) {
    _Float16* x16 = (_Float16*)d_ws;
    _Float16* w16 = (_Float16*)((char*)d_ws + xBytes);
    float*    u   = (float*)((char*)d_ws + xBytes + wBytes);

    qlora_cvt_x<<<dim3((unsigned)((size_t)M * K_TOTAL / 8 / 256)), blk, 0, stream>>>(x, x16);
    qlora_dequant_w<<<dim3((unsigned)((size_t)N_TOTAL * K_TOTAL / 8 / 256)), blk, 0, stream>>>(
        qw, scales, zeros, w16);
    qlora_lora_u<<<dim3((unsigned)((M * RANK + 255) / 256)), blk, 0, stream>>>(x, loraA, u, M);
    qlora_gemm_async<<<gGemm, blk, 0, stream>>>(x16, w16, loraB, bias, u, out);
  } else {
    float* u = (float*)d_ws;
    qlora_lora_u<<<dim3((unsigned)((M * RANK + 255) / 256)), blk, 0, stream>>>(x, loraA, u, M);
    qlora_gemm_fused<<<gGemm, blk, 0, stream>>>(x, qw, scales, zeros, loraB, bias, u, out);
  }
}
